// MyModel_44822278701234
// MI455X (gfx1250) — matrix operand order repaired and checked by reading, not compiled, not run on hardware
//
#include <hip/hip_runtime.h>
#include <hip/hip_bf16.h>

typedef __attribute__((ext_vector_type(16))) _Float16 v16h;
typedef __attribute__((ext_vector_type(8)))  _Float16 v8h;
typedef __attribute__((ext_vector_type(8)))  float    v8f;
typedef __attribute__((ext_vector_type(4)))  unsigned int u32x4;
typedef __attribute__((ext_vector_type(8)))  int i32x8;
typedef __attribute__((ext_vector_type(4)))  int i32x4;

#define TILE_W   64
#define ROWS_PB  2                       // output rows per block
#define ACT_ROWS (ROWS_PB + 2)           // halo
#define ACT_COLS 66                      // TILE_W + 2 halo
#define ACT_N    (ACT_ROWS * ACT_COLS * 32)   // 8448 halves
#define WT_N     (32 * 288)                   // 9216 halves = 18432 B

#if defined(__has_builtin)
#if __has_builtin(__builtin_amdgcn_tensor_load_to_lds)
#define HAVE_TDM 1
#endif
#endif

// ---------------------------------------------------------------------------
// one-shot: weights f32 [32,288] -> f16 in workspace
// ---------------------------------------------------------------------------
__global__ __launch_bounds__(256) void wcvt_kernel(
    const float* __restrict__ w, _Float16* __restrict__ wh)
{
    const int i = blockIdx.x * 256 + threadIdx.x;
    if (i < WT_N) wh[i] = (_Float16)w[i];
}

// ---------------------------------------------------------------------------
// conv0: 3x3 SAME, C=3 -> F=32, bias + relu.  VALU f32 accumulate (K=27,
// ~4% of FLOPs), f16 output to feed the WMMA layers.
// ---------------------------------------------------------------------------
__global__ __launch_bounds__(256) void conv0_kernel(
    const float* __restrict__ x, const float* __restrict__ k0,
    const float* __restrict__ b0, _Float16* __restrict__ out)
{
    __shared__ float kLds[27 * 32];
    __shared__ float bLds[32];
    const int tid = threadIdx.x;
    for (int i = tid; i < 27 * 32; i += 256) kLds[i] = k0[i];
    if (tid < 32) bLds[tid] = b0[tid];
    __syncthreads();

    size_t idx = (size_t)blockIdx.x * 256 + tid;     // pixel index
    const int wp = (int)(idx & 127);
    const int hp = (int)((idx >> 7) & 127);
    const int b  = (int)(idx >> 14);

    float acc[32];
    #pragma unroll
    for (int f = 0; f < 32; ++f) acc[f] = bLds[f];

    #pragma unroll
    for (int kh = 0; kh < 3; ++kh) {
        const int gh = hp + kh - 1;
        if (gh < 0 || gh >= 128) continue;
        #pragma unroll
        for (int kw = 0; kw < 3; ++kw) {
            const int gw = wp + kw - 1;
            if (gw < 0 || gw >= 128) continue;
            const float* px = x + (((size_t)b * 128 + gh) * 128 + gw) * 3;
            #pragma unroll
            for (int c = 0; c < 3; ++c) {
                const float xv = px[c];
                const float* kr = kLds + ((kh * 3 + kw) * 3 + c) * 32;
                #pragma unroll
                for (int f = 0; f < 32; ++f) acc[f] += xv * kr[f];
            }
        }
    }
    _Float16* po = out + idx * 32;
    #pragma unroll
    for (int f = 0; f < 32; ++f) {
        const float v = acc[f];
        po[f] = (_Float16)(v > 0.f ? v : 0.f);
    }
}

// ---------------------------------------------------------------------------
// my_conv2d on WMMA:  out[p,f] = relu(sum_k patch[p,k]*w[f,k] + b[f])
// K = 288 = 9 taps * 32 channels; each 32-wide k-block == one (kh,kw) tap.
// Block = 256 threads = 8 waves: 2 rows x 64 pixels x 32 filters.
// Wave (rr,pg): 16-pixel tile, BOTH 16-filter tiles (A reused), 18 WMMAs.
// Weights staged by the Tensor Data Mover (wave 0 issues tensor_load_to_lds,
// overlapped with manual halo-aware activation staging by all waves).
// ---------------------------------------------------------------------------
__global__ __launch_bounds__(256) void conv_wmma_kernel(
    const _Float16* __restrict__ in, const _Float16* __restrict__ wh,
    const float* __restrict__ bias, _Float16* __restrict__ out,
    int H, int W)
{
    __shared__ __align__(16) _Float16 actLds[ACT_N];  // [row][col 0..65][ch 0..31]
    __shared__ __align__(16) _Float16 wLds[WT_N];     // [f 0..31][k 0..287]

    const int tid = threadIdx.x;
    const int w0  = blockIdx.x * TILE_W;
    const int h0  = blockIdx.y * ROWS_PB;
    const int bb  = blockIdx.z;

#ifdef HAVE_TDM
    // --- TDM weight staging: wave 0 issues one DMA of 18432 B (2304 x 8B). ---
    if (tid < 32) {
        const unsigned int lds_addr = (unsigned int)(uintptr_t)(&wLds[0]);
        const unsigned long long ga = (unsigned long long)(uintptr_t)wh;
        u32x4 g0;
        g0[0] = 1u;                                   // count=1, user descriptor
        g0[1] = lds_addr;                             // lds_addr [63:32]
        g0[2] = (unsigned int)ga;                     // global_addr lo
        g0[3] = ((unsigned int)(ga >> 32) & 0x01FFFFFFu) | 0x80000000u; // hi + type=2
        i32x8 g1;
        g1[0] = 0x00030000;                           // data_size=3 (8B), mask=0
        g1[1] = (int)(2304u << 16);                   // tensor_dim0 = 2304 (lo16)
        g1[2] = (int)(1u << 16);                      // tensor_dim0 hi=0, tensor_dim1=1
        g1[3] = (int)(2304u << 16);                   // tensor_dim1 hi=0, tile_dim0=2304
        g1[4] = 1;                                    // tile_dim1=1, tile_dim2=0
        g1[5] = 2304;                                 // tensor_dim0_stride lo
        g1[6] = 0;                                    // stride hi, dim1_stride lo
        g1[7] = 0;
        const i32x4 gz = {0, 0, 0, 0};
#if __clang_major__ >= 23
        const i32x8 gz8 = {0, 0, 0, 0, 0, 0, 0, 0};
        __builtin_amdgcn_tensor_load_to_lds(g0, g1, gz, gz, gz8, 0);
#else
        __builtin_amdgcn_tensor_load_to_lds(g0, g1, gz, gz, 0);
#endif
    }
#else
    // Fallback: bulk 16-byte vector copies.
    {
        const uint4* ws = (const uint4*)wh;
        uint4* wd = (uint4*)wLds;
        for (int i = tid; i < WT_N / 8; i += 256) wd[i] = ws[i];
    }
#endif

    // --- Halo-aware activation staging (16B chunks; 4 chunks per pixel). ---
    {
        uint4* ad = (uint4*)actLds;
        const uint4 zv = {0u, 0u, 0u, 0u};
        const int NCH = ACT_ROWS * ACT_COLS * 4;      // 1056 chunks
        for (int i = tid; i < NCH; i += 256) {
            const int r   = i / (ACT_COLS * 4);
            const int rem = i - r * (ACT_COLS * 4);
            const int cc  = rem >> 2;
            const int c8  = rem & 3;
            const int gh  = h0 + r - 1;
            const int gw  = w0 + cc - 1;
            uint4 v = zv;
            if (gh >= 0 && gh < H && gw >= 0 && gw < W)
                v = *(const uint4*)(in + ((((size_t)bb * H + gh) * W + gw) * 32 + c8 * 8));
            ad[i] = v;
        }
    }

#ifdef HAVE_TDM
#if defined(__has_builtin) && __has_builtin(__builtin_amdgcn_s_wait_tensorcnt)
    __builtin_amdgcn_s_wait_tensorcnt(0);
#else
    asm volatile("s_wait_tensorcnt 0x0" ::: "memory");
#endif
#endif
    __syncthreads();

    const int lane = tid & 31;
    const int wv   = tid >> 5;        // wave 0..7
    const int rr   = wv >> 2;         // output row within block: 0..1
    const int pg   = wv & 3;          // pixel group 0..3 (16 pixels each)
    const int hf   = lane >> 4;       // half-wave id
    const int ln   = lane & 15;

    v8f acc0 = {};                    // filters 0..15
    v8f acc1 = {};                    // filters 16..31
    #pragma unroll
    for (int kb = 0; kb < 9; ++kb) {
        const int kh = kb / 3;
        const int kw = kb - kh * 3;

        // A fragment: row M=ln, tap (kh,kw); channel split {0..15}/{16..31}
        const _Float16* ab = actLds +
            (((kh + rr) * ACT_COLS + pg * 16 + ln + kw) * 32 + hf * 16);
        const v8h a0 = *(const v8h*)ab;
        const v8h a1 = *(const v8h*)(ab + 8);
        const v16h A = __builtin_shufflevector(a0, a1,
            0, 1, 2, 3, 4, 5, 6, 7, 8, 9, 10, 11, 12, 13, 14, 15);

        // B fragments: column N = ln (+16), K = kb*32 + hf*16 .. +15
        const _Float16* bp0 = wLds + (ln * 288 + kb * 32 + hf * 16);
        const v8h b00 = *(const v8h*)bp0;
        const v8h b01 = *(const v8h*)(bp0 + 8);
        const v16h B0 = __builtin_shufflevector(b00, b01,
            0, 1, 2, 3, 4, 5, 6, 7, 8, 9, 10, 11, 12, 13, 14, 15);

        const _Float16* bp1 = wLds + ((16 + ln) * 288 + kb * 32 + hf * 16);
        const v8h b10 = *(const v8h*)bp1;
        const v8h b11 = *(const v8h*)(bp1 + 8);
        const v16h B1 = __builtin_shufflevector(b10, b11,
            0, 1, 2, 3, 4, 5, 6, 7, 8, 9, 10, 11, 12, 13, 14, 15);

        acc0 = __builtin_amdgcn_wmma_f32_16x16x32_f16(
            false, A, false, B0, (short)0, acc0, false, false);
        acc1 = __builtin_amdgcn_wmma_f32_16x16x32_f16(
            false, A, false, B1, (short)0, acc1, false, false);
    }

    // D layout: VGPR r, lane l -> M = r + 8*(l>=16), N = l&15. Bias+relu, f16 out.
    const float bv0 = bias[ln];
    const float bv1 = bias[16 + ln];
    const int h = h0 + rr;
    #pragma unroll
    for (int r = 0; r < 8; ++r) {
        const int p = pg * 16 + r + hf * 8;
        const size_t base = (((size_t)bb * H + h) * W + (w0 + p)) * 32;
        float v0 = acc0[r] + bv0;
        float v1 = acc1[r] + bv1;
        out[base + ln]      = (_Float16)(v0 > 0.f ? v0 : 0.f);
        out[base + 16 + ln] = (_Float16)(v1 > 0.f ? v1 : 0.f);
    }
}

// ---------------------------------------------------------------------------
// 2x2 max pool, stride 2 (f16 -> f16)
// ---------------------------------------------------------------------------
__global__ __launch_bounds__(256) void pool_kernel(
    const _Float16* __restrict__ in, _Float16* __restrict__ out, int H, int W)
{
    const int OH = H / 2, OW = W / 2;
    const size_t total = (size_t)16 * OH * OW * 32;
    size_t idx = (size_t)blockIdx.x * 256 + threadIdx.x;
    if (idx >= total) return;
    const int c = (int)(idx & 31);
    size_t t = idx >> 5;
    const int ow = (int)(t % OW); t /= OW;
    const int oh = (int)(t % OH); t /= OH;
    const int b  = (int)t;
    const _Float16* base = in + (((size_t)b * H + oh * 2) * W + ow * 2) * 32 + c;
    const float v0 = (float)base[0];
    const float v1 = (float)base[32];
    const float v2 = (float)base[(size_t)W * 32];
    const float v3 = (float)base[(size_t)W * 32 + 32];
    out[idx] = (_Float16)fmaxf(fmaxf(v0, v1), fmaxf(v2, v3));
}

// ---------------------------------------------------------------------------
// dense [16,32768](f16) @ Wd [32768,10](f32) + bd, softmax -> f32 out
// ---------------------------------------------------------------------------
__global__ __launch_bounds__(256) void dense_softmax_kernel(
    const _Float16* __restrict__ flat, const float* __restrict__ Wd,
    const float* __restrict__ bd, float* __restrict__ out)
{
    __shared__ float red[10][256];
    const int b = blockIdx.x;
    const int t = threadIdx.x;

    float acc[10];
    #pragma unroll
    for (int j = 0; j < 10; ++j) acc[j] = 0.f;

    const _Float16* fb = flat + (size_t)b * 32768;
    for (int k = t; k < 32768; k += 256) {
        const float v = (float)fb[k];
        const float* wr = Wd + (size_t)k * 10;
        #pragma unroll
        for (int j = 0; j < 10; ++j) acc[j] += v * wr[j];
    }
    #pragma unroll
    for (int j = 0; j < 10; ++j) red[j][t] = acc[j];
    __syncthreads();
    for (int s = 128; s > 0; s >>= 1) {
        if (t < s) {
            #pragma unroll
            for (int j = 0; j < 10; ++j) red[j][t] += red[j][t + s];
        }
        __syncthreads();
    }
    if (t == 0) {
        float logits[10];
        float mx = -1e30f;
        #pragma unroll
        for (int j = 0; j < 10; ++j) {
            logits[j] = red[j][0] + bd[j];
            mx = fmaxf(mx, logits[j]);
        }
        float s = 0.f;
        #pragma unroll
        for (int j = 0; j < 10; ++j) {
            logits[j] = __expf(logits[j] - mx);
            s += logits[j];
        }
        #pragma unroll
        for (int j = 0; j < 10; ++j) out[b * 10 + j] = logits[j] / s;
    }
}

// ---------------------------------------------------------------------------
extern "C" void kernel_launch(void* const* d_in, const int* in_sizes, int n_in,
                              void* d_out, int out_size, void* d_ws, size_t ws_size,
                              hipStream_t stream)
{
    const float* x  = (const float*)d_in[0];   // [16,128,128,3]
    const float* k0 = (const float*)d_in[1];   // [3,3,3,32]
    const float* b0 = (const float*)d_in[2];   // [32]
    const float* w  = (const float*)d_in[3];   // [32,288]
    const float* bc = (const float*)d_in[4];   // [32]
    const float* Wd = (const float*)d_in[5];   // [32768,10]
    const float* bd = (const float*)d_in[6];   // [10]
    float* out = (float*)d_out;                // [16,10]

    const size_t NACT = (size_t)16 * 128 * 128 * 32;
    _Float16* wH   = (_Float16*)d_ws;          // 9216 halves (reserve 16384)
    _Float16* bufA = wH + 16384;
    _Float16* bufB = bufA + NACT;

    wcvt_kernel<<<(WT_N + 255) / 256, 256, 0, stream>>>(w, wH);

    conv0_kernel<<<(16 * 128 * 128) / 256, 256, 0, stream>>>(x, k0, b0, bufA);

    conv_wmma_kernel<<<dim3(128 / TILE_W, 128 / ROWS_PB, 16), 256, 0, stream>>>(
        bufA, wH, bc, bufB, 128, 128);
    conv_wmma_kernel<<<dim3(128 / TILE_W, 128 / ROWS_PB, 16), 256, 0, stream>>>(
        bufB, wH, bc, bufA, 128, 128);

    {
        const size_t n = (size_t)16 * 64 * 64 * 32;
        pool_kernel<<<(unsigned)((n + 255) / 256), 256, 0, stream>>>(bufA, bufB, 128, 128);
    }

    conv_wmma_kernel<<<dim3(64 / TILE_W, 64 / ROWS_PB, 16), 256, 0, stream>>>(
        bufB, wH, bc, bufA, 64, 64);
    conv_wmma_kernel<<<dim3(64 / TILE_W, 64 / ROWS_PB, 16), 256, 0, stream>>>(
        bufA, wH, bc, bufB, 64, 64);

    {
        const size_t n = (size_t)16 * 32 * 32 * 32;
        pool_kernel<<<(unsigned)((n + 255) / 256), 256, 0, stream>>>(bufB, bufA, 64, 64);
    }

    dense_softmax_kernel<<<16, 256, 0, stream>>>(bufA, Wd, bd, out);
}
